// GGNNLayer_4698694222084
// MI455X (gfx1250) — matrix operand order repaired and checked by reading, not compiled, not run on hardware
//
#include <hip/hip_runtime.h>
#include <hip/hip_bf16.h>

// ---------------------------------------------------------------------------
// GGNN layer for MI455X (gfx1250, wave32, WMMA).
//   Phase 1: bin edges by type (histogram/prefix/compact, atomics).
//   Phase 2: repack all 12 weight matrices (6 edge-type slices of W_edge +
//            6 GRU matrices) into bf16, pre-swizzled into the WMMA B-fragment
//            layout so GEMM inner loops are contiguous b128 loads.
//   Phase 3: per-type GEMM [M_t,512]x[512,512] with v_wmma_f32_16x16x32_bf16,
//            A rows gathered via source_indices with GLOBAL_LOAD_ASYNC_TO_LDS
//            (ASYNCcnt), converted fp32->bf16 once in LDS, C tiles
//            scatter-added into proposed (= d_out) with f32 atomics
//            (proposed fits in the 192MB L2, so atomics stay on-chip).
//   Phase 4: fused GRU: 6 WMMA GEMMs + sigmoid/tanh elementwise per 16-row
//            tile, async-staged A rows, overwriting d_out in place.
// ---------------------------------------------------------------------------

#define N_NODES 50000
#define N_EDGES 100000
#define HID     512
#define NTYPE   6
#define MAT_ELEMS (HID * HID)          // 262144 elements per packed matrix

// workspace layout (bytes)
#define CNT_OFF 0                       // int[48]: [0..5]=count [16..21]=cursor [32..37]=offset
#define BIN_OFF 1024                    // int[N_EDGES]
#define PK_OFF  401408                  // __bf16[12 * MAT_ELEMS]  (~6.3 MB)

typedef __attribute__((ext_vector_type(16))) __bf16 v16bf;
typedef __attribute__((ext_vector_type(8)))  float  v8f;

// ---------------- gfx1250 async copy to LDS ----------------
// Low 32 bits of a flat pointer into LDS are the LDS byte offset
// (see ISA 10.2, LDS aperture).
__device__ __forceinline__ void async_ld_b128(const float* gptr, float* lptr) {
  unsigned lds = (unsigned)(uintptr_t)lptr;
  unsigned long long ga = (unsigned long long)(uintptr_t)gptr;
  asm volatile("global_load_async_to_lds_b128 %0, %1, off"
               :: "v"(lds), "v"(ga) : "memory");
}
__device__ __forceinline__ void wait_async0() {
  asm volatile("s_wait_asynccnt 0x0" ::: "memory");
}

// ---------------- fragment loaders ----------------

// A fragment (16x32 bf16) from LDS tile As[16][512], K offset k0.
// Contiguous K order used here (not the ISA table order): lane m=lane&15; khalf=lane>>4; elems 0..15 -> K=khalf*16+i
// (matches packed B). Two contiguous 16B runs -> 2x ds_load_b128.
__device__ __forceinline__ v16bf load_a_frag(const __bf16* __restrict__ As,
                                             int lane, int k0) {
  int m = lane & 15;
  int kh = (lane >> 4) << 4;
  const __bf16* p0 = As + m * HID + k0 + kh;
  const __bf16* p1 = p0 + 8;
  v16bf a;
#pragma unroll
  for (int i = 0; i < 8; ++i) a[i] = p0[i];
#pragma unroll
  for (int i = 0; i < 8; ++i) a[8 + i] = p1[i];
  return a;
}

// B fragment (32x16 bf16) from the pre-swizzled packed buffer:
// packed[( nt*16 + kt )*512 + lane*16 + i] == B[kt*32 + (lane>>4)*16 + i][nt*16 + (lane&15)]
__device__ __forceinline__ v16bf load_b_frag(const __bf16* __restrict__ Wpk,
                                             int nt, int kt, int lane) {
  const __bf16* p = Wpk + ((size_t)(nt * 16 + kt) * HID) + lane * 16;
  v16bf b;
#pragma unroll
  for (int i = 0; i < 16; ++i) b[i] = p[i];
  return b;
}

// full K=512 accumulation of one 16x16 tile: 16 x v_wmma_f32_16x16x32_bf16
__device__ __forceinline__ v8f mm_tile(const __bf16* __restrict__ As,
                                       const __bf16* __restrict__ Wpk,
                                       int nt, int lane, v8f acc) {
#pragma unroll
  for (int kt = 0; kt < 16; ++kt) {
    if (kt + 1 < 16)
      __builtin_prefetch(Wpk + ((size_t)(nt * 16 + kt + 1) * HID) + lane * 16,
                         0, 3);
    v16bf a = load_a_frag(As, lane, kt * 32);
    v16bf b = load_b_frag(Wpk, nt, kt, lane);
    acc = __builtin_amdgcn_wmma_f32_16x16x32_bf16(false, a, false, b,
                                                  (short)0, acc, false, false);
  }
  return acc;
}

__device__ __forceinline__ float sigmoidf_(float x) {
  return 1.f / (1.f + __expf(-x));
}
__device__ __forceinline__ float tanhf_(float x) {
  x = fminf(fmaxf(x, -15.f), 15.f);
  float e = __expf(2.f * x);
  return (e - 1.f) / (e + 1.f);
}

// ---------------- phase 0: zero output + counters ----------------
__global__ __launch_bounds__(256) void k_zero(float* __restrict__ out,
                                              int* __restrict__ counters) {
  size_t i = (size_t)blockIdx.x * blockDim.x + threadIdx.x;
  size_t tot4 = (size_t)N_NODES * HID / 4;
  if (i < tot4) ((float4*)out)[i] = make_float4(0.f, 0.f, 0.f, 0.f);
  if (blockIdx.x == 0 && threadIdx.x < 48) counters[threadIdx.x] = 0;
}

// ---------------- phase 1: bin edges by type ----------------
__global__ __launch_bounds__(256) void k_count(const int* __restrict__ types,
                                               int* __restrict__ counters) {
  int e = blockIdx.x * blockDim.x + threadIdx.x;
  if (e < N_EDGES) atomicAdd(&counters[types[e]], 1);
}

__global__ void k_prefix(int* __restrict__ counters) {
  int run = 0;
  for (int t = 0; t < NTYPE; ++t) {
    counters[32 + t] = run;   // offset
    counters[16 + t] = run;   // cursor
    run += counters[t];
  }
}

__global__ __launch_bounds__(256) void k_fill(const int* __restrict__ types,
                                              int* __restrict__ counters,
                                              int* __restrict__ binned) {
  int e = blockIdx.x * blockDim.x + threadIdx.x;
  if (e < N_EDGES) {
    int t = types[e];
    int pos = atomicAdd(&counters[16 + t], 1);
    binned[pos] = e;
  }
}

// ---------------- phase 2: weight repack into WMMA-B layout ----------------
__global__ __launch_bounds__(256) void k_repack(
    const float* __restrict__ W_edge,
    const float* __restrict__ Wir, const float* __restrict__ Wiz,
    const float* __restrict__ Win, const float* __restrict__ Whr,
    const float* __restrict__ Whz, const float* __restrict__ Whn,
    __bf16* __restrict__ pk) {
  int idx = blockIdx.x * 256 + threadIdx.x;        // 0 .. 12*262144-1
  int mat = idx >> 18;
  int r   = idx & (MAT_ELEMS - 1);
  int i    = r & 15;
  int lane = (r >> 4) & 31;
  int kt   = (r >> 9) & 15;
  int nt   = r >> 13;
  int n = nt * 16 + (lane & 15);
  int k = kt * 32 + ((lane >> 4) * 16 + i);
  float v;
  if (mat < NTYPE) {
    v = W_edge[(size_t)k * (NTYPE * HID) + mat * HID + n];
  } else {
    const float* W = (mat == 6) ? Wir : (mat == 7) ? Wiz : (mat == 8) ? Win
                   : (mat == 9) ? Whr : (mat == 10) ? Whz : Whn;
    v = W[(size_t)k * HID + n];
  }
  pk[idx] = (__bf16)v;
}

// ---------------- phase 3: per-type edge GEMM + atomic scatter ----------------
__global__ __launch_bounds__(256) void k_edge_msgs(
    const float* __restrict__ emb, const int* __restrict__ src_idx,
    const int* __restrict__ dst_idx, const int* __restrict__ binned,
    const int* __restrict__ counters, const float* __restrict__ b_edge,
    const __bf16* __restrict__ pk, float* __restrict__ proposed) {
  int t = blockIdx.y;
  int cnt = counters[t];
  int off = counters[32 + t];
  int mbase = blockIdx.x * 16;
  if (mbase >= cnt) return;

  __shared__ float  Araw[16 * HID];   // 32 KB, async-staged fp32
  __shared__ __bf16 As[16 * HID];     // 16 KB
  __shared__ int drow[16];

  int tid = threadIdx.x;
  int row = tid >> 4;
  int cseg = (tid & 15) * 32;
  int er = mbase + row;
  bool valid = (er < cnt);
  int srow = 0, dr = -1;
  if (valid) {
    int e = binned[off + er];
    srow = src_idx[e];
    dr = dst_idx[e];
  }
  const float* prow = emb + (size_t)srow * HID;
  if (valid) {
#pragma unroll
    for (int j = 0; j < 32; j += 4)
      async_ld_b128(prow + cseg + j, &Araw[row * HID + cseg + j]);
  }
  if ((tid & 15) == 0) drow[row] = dr;
  wait_async0();
  __syncthreads();
#pragma unroll
  for (int j = 0; j < 32; ++j) {
    int o = row * HID + cseg + j;
    As[o] = valid ? (__bf16)Araw[o] : (__bf16)0.f;
  }
  __syncthreads();

  int wave = tid >> 5, lane = tid & 31;
  const __bf16* Wt = pk + (size_t)t * MAT_ELEMS;
  int n_local = lane & 15;
  int mhi = (lane >> 4) * 8;

#pragma unroll
  for (int st = 0; st < 4; ++st) {
    int nt = wave * 4 + st;
    v8f acc = {0.f, 0.f, 0.f, 0.f, 0.f, 0.f, 0.f, 0.f};
    acc = mm_tile(As, Wt, nt, lane, acc);
    int col = nt * 16 + n_local;
    float bias = b_edge[t * HID + col];
#pragma unroll
    for (int v = 0; v < 8; ++v) {
      int d = drow[mhi + v];
      if (d >= 0)
        atomicAdd(proposed + (size_t)d * HID + col, acc[v] + bias);
    }
  }
}

// ---------------- phase 4: fused GRU (6 GEMMs + elementwise) ----------------
__global__ __launch_bounds__(256) void k_gru(
    const float* __restrict__ emb, float* __restrict__ inout,
    const __bf16* __restrict__ pk, const float* __restrict__ b_ir,
    const float* __restrict__ b_iz, const float* __restrict__ b_in,
    const float* __restrict__ b_hn) {
  __shared__ float  Praw[16 * HID];   // 32 KB, async-staged proposed rows
  __shared__ float  Hraw[16 * HID];   // 32 KB, async-staged h rows (also Hf)
  __shared__ __bf16 Ps[16 * HID];     // 16 KB
  __shared__ __bf16 Hs[16 * HID];     // 16 KB

  int tid = threadIdx.x;
  int mbase = blockIdx.x * 16;
  int row = tid >> 4;
  int cseg = (tid & 15) * 32;
  const float* prow = inout + (size_t)(mbase + row) * HID;
  const float* hrow = emb + (size_t)(mbase + row) * HID;
#pragma unroll
  for (int j = 0; j < 32; j += 4) {
    async_ld_b128(prow + cseg + j, &Praw[row * HID + cseg + j]);
    async_ld_b128(hrow + cseg + j, &Hraw[row * HID + cseg + j]);
  }
  wait_async0();
  __syncthreads();
#pragma unroll
  for (int j = 0; j < 32; ++j) {
    int o = row * HID + cseg + j;
    Ps[o] = (__bf16)Praw[o];
    Hs[o] = (__bf16)Hraw[o];
  }
  __syncthreads();

  int wave = tid >> 5, lane = tid & 31;
  const __bf16* Wir = pk + (size_t)6  * MAT_ELEMS;
  const __bf16* Wiz = pk + (size_t)7  * MAT_ELEMS;
  const __bf16* Win = pk + (size_t)8  * MAT_ELEMS;
  const __bf16* Whr = pk + (size_t)9  * MAT_ELEMS;
  const __bf16* Whz = pk + (size_t)10 * MAT_ELEMS;
  const __bf16* Whn = pk + (size_t)11 * MAT_ELEMS;
  int n_local = lane & 15;
  int mhi = (lane >> 4) * 8;
  const v8f vzero = {0.f, 0.f, 0.f, 0.f, 0.f, 0.f, 0.f, 0.f};

  for (int st = 0; st < 4; ++st) {
    int nt = wave * 4 + st;
    int col = nt * 16 + n_local;

    v8f accR = mm_tile(Ps, Wir, nt, lane, vzero);
    accR     = mm_tile(Hs, Whr, nt, lane, accR);
    v8f accZ = mm_tile(Ps, Wiz, nt, lane, vzero);
    accZ     = mm_tile(Hs, Whz, nt, lane, accZ);
    v8f accHN = mm_tile(Hs, Whn, nt, lane, vzero);
    v8f accN  = mm_tile(Ps, Win, nt, lane, vzero);

    float bir = b_ir[col], biz = b_iz[col], bin_ = b_in[col], bhn = b_hn[col];
#pragma unroll
    for (int v = 0; v < 8; ++v) {
      float rg = sigmoidf_(accR[v] + bir);
      float zg = sigmoidf_(accZ[v] + biz);
      float ng = tanhf_(accN[v] + bin_ + rg * (accHN[v] + bhn));
      float h = Hraw[(mhi + v) * HID + col];
      inout[(size_t)(mbase + mhi + v) * HID + col] = (1.f - zg) * ng + zg * h;
    }
  }
}

// ---------------------------------------------------------------------------
extern "C" void kernel_launch(void* const* d_in, const int* in_sizes, int n_in,
                              void* d_out, int out_size, void* d_ws,
                              size_t ws_size, hipStream_t stream) {
  const float* emb    = (const float*)d_in[0];
  const int*   src    = (const int*)d_in[1];
  const int*   dst    = (const int*)d_in[2];
  const int*   types  = (const int*)d_in[3];
  const float* W_edge = (const float*)d_in[6];
  const float* b_edge = (const float*)d_in[7];
  const float* W_ir   = (const float*)d_in[8];
  const float* W_iz   = (const float*)d_in[9];
  const float* W_in   = (const float*)d_in[10];
  const float* b_ir   = (const float*)d_in[11];
  const float* b_iz   = (const float*)d_in[12];
  const float* b_in   = (const float*)d_in[13];
  const float* W_hr   = (const float*)d_in[14];
  const float* W_hz   = (const float*)d_in[15];
  const float* W_hn   = (const float*)d_in[16];
  const float* b_hn   = (const float*)d_in[17];

  float* out = (float*)d_out;   // doubles as `proposed` scratch
  char* ws = (char*)d_ws;
  int*    counters = (int*)(ws + CNT_OFF);
  int*    binned   = (int*)(ws + BIN_OFF);
  __bf16* pk       = (__bf16*)(ws + PK_OFF);

  // phase 0: zero proposed (in d_out) and counters
  int zgrid = (int)(((size_t)N_NODES * HID / 4 + 255) / 256);
  k_zero<<<zgrid, 256, 0, stream>>>(out, counters);

  // phase 1: histogram / prefix / compaction
  int egrid = (N_EDGES + 255) / 256;
  k_count<<<egrid, 256, 0, stream>>>(types, counters);
  k_prefix<<<1, 1, 0, stream>>>(counters);
  k_fill<<<egrid, 256, 0, stream>>>(types, counters, binned);

  // phase 2: weight repack (12 matrices -> bf16 WMMA-B layout)
  k_repack<<<12 * MAT_ELEMS / 256, 256, 0, stream>>>(
      W_edge, W_ir, W_iz, W_in, W_hr, W_hz, W_hn, pk);

  // phase 3: per-type edge GEMMs, scatter into proposed
  dim3 eg((N_EDGES + 15) / 16, NTYPE);
  k_edge_msgs<<<eg, 256, 0, stream>>>(emb, src, dst, binned, counters, b_edge,
                                      pk, out);

  // phase 4: fused GRU, overwrite d_out in place
  k_gru<<<N_NODES / 16, 256, 0, stream>>>(emb, out, pk, b_ir, b_iz, b_in,
                                          b_hn);
}
